// CollaborativeGPTwithItemLMHeadBatch_31688268710409
// MI455X (gfx1250) — matrix operand order repaired and checked by reading, not compiled, not run on hardware
//
#include <hip/hip_runtime.h>
#include <math.h>

typedef __bf16 bf16;
typedef __bf16 v16bf __attribute__((ext_vector_type(16)));
typedef float  v8f   __attribute__((ext_vector_type(8)));
typedef int    v2i_  __attribute__((vector_size(2 * sizeof(int))));

#define NUM_ITEMS   100000
#define D_DIM       768
#define B_SZ        8
#define L_MAIN      128
#define L_TOT       160          // L_PROMPT + L_MAIN
#define LABEL_OFF   161669       // 151669 + 10000
#define NROWS       1024         // B*L_MAIN
#define NTILES      782          // ceil(100000/128)
#define NPAD_ITEMS  (NTILES * 128)   // 100096: padded rows in split planes
#define BK          64           // K-chunk staged in LDS (2 WMMA K-steps)
#define STRH        72           // bf16 row stride (144B): b128-aligned, conflict-free

// dynamic-LDS layout (bf16 elements) for the double-buffered presplit kernel
#define A_ELEMS     (16 * STRH)       // 1152
#define B_ELEMS     (128 * STRH)      // 9216
#define OFF_ALO     A_ELEMS           // 1152
#define OFF_BHI     (2 * A_ELEMS)     // 2304
#define OFF_BLO     (2 * A_ELEMS + B_ELEMS)   // 11520
#define BUF_ELEMS   (2 * A_ELEMS + 2 * B_ELEMS) // 20736 elems = 41472 B
#define COMB_OFF_B  (2 * BUF_ELEMS * 2)       // 82944 bytes
#define SMEM_BYTES  (COMB_OFF_B + 128 * 8)    // + comb float2[128] = 83968

#define ISSUES_PER_CHUNK 18   // per-wave async ops per chunk (2 A + 16 B)

#define NEG_INF (-__builtin_inff())

#if defined(__has_builtin)
#if __has_builtin(__builtin_amdgcn_global_load_async_to_lds_b64) && \
    __has_builtin(__builtin_amdgcn_s_wait_asynccnt)
#define HAVE_ASYNC_LDS 1
#endif
#endif

static __device__ __forceinline__ void stage_b64(const void* g, void* l) {
#ifdef HAVE_ASYNC_LDS
    v2i_* gp = (v2i_*)(void*)g;
    v2i_* lp = (v2i_*)l;
    __builtin_amdgcn_global_load_async_to_lds_b64(
        (__attribute__((address_space(1))) v2i_*)gp,
        (__attribute__((address_space(3))) v2i_*)lp, 0, 0);
#else
    *(uint2*)l = *(const uint2*)g;
#endif
}

static __device__ __forceinline__ unsigned pack_bf16x2(bf16 a, bf16 b) {
    unsigned short ua = __builtin_bit_cast(unsigned short, a);
    unsigned short ub = __builtin_bit_cast(unsigned short, b);
    return (unsigned)ua | ((unsigned)ub << 16);
}

// ---------------------------------------------------------------------------
// Kernel 0: one-shot fp32 -> packed (hi,lo) bf16 plane split (memory bound).
// hi = rne_bf16(x); lo = rne_bf16(x - hi).  Rows >= nvalid are zero-padded so
// the GEMM staging needs no bounds checks.
// ---------------------------------------------------------------------------
__global__ __launch_bounds__(256)
void split_planes_kernel(const float* __restrict__ src,
                         unsigned* __restrict__ hi, unsigned* __restrict__ lo,
                         long nvalid, long ntotal)
{
    for (long i = (long)blockIdx.x * 256 + threadIdx.x; i < ntotal;
         i += (long)gridDim.x * 256) {
        if (i < nvalid) {
            float2 v = ((const float2*)src)[i];
            bf16 h0 = (bf16)v.x, h1 = (bf16)v.y;
            bf16 l0 = (bf16)(v.x - (float)h0), l1 = (bf16)(v.y - (float)h1);
            hi[i] = pack_bf16x2(h0, h1);
            lo[i] = pack_bf16x2(l0, l1);
        } else {
            hi[i] = 0u;
            lo[i] = 0u;
        }
    }
}

// ---------------------------------------------------------------------------
// Kernel 1 (primary): fused logits GEMM + online softmax stats.
// Double-buffered async staging from pre-split bf16 planes; each K=32 step is
// 3x v_wmma_f32_16x16x32_bf16 (hi*hi + hi*lo + lo*hi) with fp32 accumulation.
// grid.x = 64 row-tiles (16 rows), grid.y = VB item-slices.
// ---------------------------------------------------------------------------
__global__ __launch_bounds__(256)
void ce_gemm_softmax_presplit(const unsigned* __restrict__ hidHi,
                              const unsigned* __restrict__ hidLo,
                              const unsigned* __restrict__ itemHi,
                              const unsigned* __restrict__ itemLo,
                              float2* __restrict__ ws, int vblocks)
{
    extern __shared__ char smem[];

    const int t    = threadIdx.x;
    const int wave = t >> 5;
    const int lane = t & 31;
    const int half = lane >> 4;
    const int ln   = lane & 15;
    const int rowBase = blockIdx.x * 16;     // global row = b*128 + l

    float Mrun[8], Srun[8];
#pragma unroll
    for (int r = 0; r < 8; ++r) { Mrun[r] = NEG_INF; Srun[r] = 0.0f; }

    const int aRow = ln * STRH;
    const int bRow = (wave * 16 + ln) * STRH;

    // issue one K-chunk of (tile_, kc) into LDS buffer p (branch-free: padded planes)
    auto issue_chunk = [&](int p, int tile_, int kc) {
        bf16* base = (bf16*)smem + p * BUF_ELEMS;
        bf16* aHi = base;
        bf16* aLo = base + OFF_ALO;
        bf16* bHi = base + OFF_BHI;
        bf16* bLo = base + OFF_BLO;
        const int v0 = tile_ * 128;
        {
            int r = t >> 4, q = t & 15;
            size_t g = (size_t)(rowBase + r) * (D_DIM / 2) + (kc >> 1) + q * 2;
            stage_b64(hidHi + g, aHi + r * STRH + q * 4);
            stage_b64(hidLo + g, aLo + r * STRH + q * 4);
        }
#pragma unroll
        for (int i = 0; i < 8; ++i) {
            int idx = t + i * 256;
            int r = idx >> 4, q = idx & 15;
            size_t g = (size_t)(v0 + r) * (D_DIM / 2) + (kc >> 1) + q * 2;
            stage_b64(itemHi + g, bHi + r * STRH + q * 4);
            stage_b64(itemLo + g, bLo + r * STRH + q * 4);
        }
    };

    int parity = 0;
    issue_chunk(0, blockIdx.y, 0);   // prologue

    for (int tile = blockIdx.y; tile < NTILES; tile += vblocks) {
        const int v0 = tile * 128;

        v8f c;
#pragma unroll
        for (int r = 0; r < 8; ++r) c[r] = 0.0f;

        for (int kc = 0; kc < D_DIM; kc += BK) {
            const bool last_chunk = (kc + BK >= D_DIM);
            const int  ntile = tile + vblocks;
            const bool has_next = !last_chunk || (ntile < NTILES);
            if (has_next) {
                issue_chunk(parity ^ 1, last_chunk ? ntile : tile,
                            last_chunk ? 0 : kc + BK);
#ifdef HAVE_ASYNC_LDS
                __builtin_amdgcn_s_wait_asynccnt(ISSUES_PER_CHUNK);
#endif
            } else {
#ifdef HAVE_ASYNC_LDS
                __builtin_amdgcn_s_wait_asynccnt(0);
#endif
            }
            __syncthreads();   // all waves' staging for buffer `parity` complete

            const bf16* base = (const bf16*)smem + parity * BUF_ELEMS;
            const bf16* AhHi = base;
            const bf16* AhLo = base + OFF_ALO;
            const bf16* BeHi = base + OFF_BHI;
            const bf16* BeLo = base + OFF_BLO;

            // two K=32 steps, 3 bf16 WMMAs each (hi*hi + hi*lo + lo*hi)
#pragma unroll
            for (int s = 0; s < 2; ++s) {
                const int k0 = 32 * s;
                v16bf ah, al, bh, bl;
                // A: lane half holds K = {16h..16h+15} contiguously
                const bf16* pa = &AhHi[aRow + k0 + 16 * half];
                __builtin_memcpy(&ah, pa, 32);
                const bf16* pal = &AhLo[aRow + k0 + 16 * half];
                __builtin_memcpy(&al, pal, 32);
                // B: lane half holds K = {16h..16h+15} contiguously
                const bf16* pb = &BeHi[bRow + k0 + 16 * half];
                __builtin_memcpy(&bh, pb, 32);
                const bf16* pbl = &BeLo[bRow + k0 + 16 * half];
                __builtin_memcpy(&bl, pbl, 32);

                c = __builtin_amdgcn_wmma_f32_16x16x32_bf16(
                        false, ah, false, bh, (short)0, c, false, false);
                c = __builtin_amdgcn_wmma_f32_16x16x32_bf16(
                        false, ah, false, bl, (short)0, c, false, false);
                c = __builtin_amdgcn_wmma_f32_16x16x32_bf16(
                        false, al, false, bh, (short)0, c, false, false);
            }
            parity ^= 1;
            __syncthreads();   // buffer reads done before it is re-staged
        }

        // tile logits in c: lane holds col n=ln, VGPR r holds row (r + 8*half)
        const int vcol = v0 + wave * 16 + ln;
        const bool valid = vcol < NUM_ITEMS;   // padded columns masked out
#pragma unroll
        for (int r = 0; r < 8; ++r) {
            float x = valid ? c[r] : NEG_INF;
            float m = x;
#pragma unroll
            for (int msk = 1; msk < 16; msk <<= 1)
                m = fmaxf(m, __shfl_xor(m, msk, 32));
            if (m > NEG_INF) {
                float e = __expf(x - m);
#pragma unroll
                for (int msk = 1; msk < 16; msk <<= 1)
                    e += __shfl_xor(e, msk, 32);
                float Mo = Mrun[r];
                float Mn = fmaxf(Mo, m);
                Srun[r] = Srun[r] * __expf(Mo - Mn) + e * __expf(m - Mn);
                Mrun[r] = Mn;
            }
        }
    }

    // combine 8 waves' (M,S) per row, write per-(row, slice) partials
    float2* comb = (float2*)(smem + COMB_OFF_B);
    if (ln == 0) {
#pragma unroll
        for (int r = 0; r < 8; ++r) {
            float2 p; p.x = Mrun[r]; p.y = Srun[r];
            comb[wave * 16 + (r + 8 * half)] = p;
        }
    }
    __syncthreads();
    if (t < 16) {
        float M = NEG_INF, S = 0.0f;
#pragma unroll
        for (int w = 0; w < 8; ++w) {
            float2 p = comb[w * 16 + t];
            if (p.x > NEG_INF) {
                float Mn = fmaxf(M, p.x);
                S = S * __expf(M - Mn) + p.y * __expf(p.x - Mn);
                M = Mn;
            }
        }
        float2 o; o.x = M; o.y = S;
        ws[(size_t)(rowBase + t) * vblocks + blockIdx.y] = o;
    }
}

// ---------------------------------------------------------------------------
// Kernel 1 (fallback, small workspace): convert fp32 -> bf16 hi/lo on the fly.
// ---------------------------------------------------------------------------
__global__ __launch_bounds__(256)
void ce_gemm_softmax_fallback(const float* __restrict__ hidden,
                              const float* __restrict__ item_emb,
                              float2* __restrict__ ws, int vblocks)
{
    __shared__ bf16 AhHi[16 * STRH],  AhLo[16 * STRH];
    __shared__ bf16 BeHi[128 * STRH], BeLo[128 * STRH];
    __shared__ float2 comb[8 * 16];

    const int t    = threadIdx.x;
    const int wave = t >> 5;
    const int lane = t & 31;
    const int half = lane >> 4;
    const int ln   = lane & 15;
    const int rowBase = blockIdx.x * 16;

    float Mrun[8], Srun[8];
#pragma unroll
    for (int r = 0; r < 8; ++r) { Mrun[r] = NEG_INF; Srun[r] = 0.0f; }

    const int aRow = ln * STRH;
    const int bRow = (wave * 16 + ln) * STRH;

    for (int tile = blockIdx.y; tile < NTILES; tile += vblocks) {
        const int v0 = tile * 128;
        v8f c;
#pragma unroll
        for (int r = 0; r < 8; ++r) c[r] = 0.0f;

        for (int kc = 0; kc < D_DIM; kc += BK) {
            __syncthreads();
#pragma unroll
            for (int i = 0; i < 2; ++i) {
                int idx = t + i * 256;
                int r = idx >> 5, cp = idx & 31, col = cp * 2;
                const float2 v = *(const float2*)(hidden + (size_t)(rowBase + r) * D_DIM + kc + col);
                bf16 h0 = (bf16)v.x, h1 = (bf16)v.y;
                bf16 l0 = (bf16)(v.x - (float)h0), l1 = (bf16)(v.y - (float)h1);
                *(unsigned*)&AhHi[r * STRH + col] = pack_bf16x2(h0, h1);
                *(unsigned*)&AhLo[r * STRH + col] = pack_bf16x2(l0, l1);
            }
#pragma unroll
            for (int i = 0; i < 16; ++i) {
                int idx = t + i * 256;
                int r = idx >> 5, cp = idx & 31, col = cp * 2;
                int v = v0 + r;
                float2 val;
                if (v < NUM_ITEMS)
                    val = *(const float2*)(item_emb + (size_t)v * D_DIM + kc + col);
                else { val.x = 0.0f; val.y = 0.0f; }
                bf16 h0 = (bf16)val.x, h1 = (bf16)val.y;
                bf16 l0 = (bf16)(val.x - (float)h0), l1 = (bf16)(val.y - (float)h1);
                *(unsigned*)&BeHi[r * STRH + col] = pack_bf16x2(h0, h1);
                *(unsigned*)&BeLo[r * STRH + col] = pack_bf16x2(l0, l1);
            }
            __syncthreads();
#pragma unroll
            for (int s = 0; s < 2; ++s) {
                const int k0 = 32 * s;
                v16bf ah, al, bh, bl;
                const bf16* pa = &AhHi[aRow + k0 + 16 * half];
                __builtin_memcpy(&ah, pa, 32);
                const bf16* pal = &AhLo[aRow + k0 + 16 * half];
                __builtin_memcpy(&al, pal, 32);
                const bf16* pb = &BeHi[bRow + k0 + 16 * half];
                __builtin_memcpy(&bh, pb, 32);
                const bf16* pbl = &BeLo[bRow + k0 + 16 * half];
                __builtin_memcpy(&bl, pbl, 32);

                c = __builtin_amdgcn_wmma_f32_16x16x32_bf16(
                        false, ah, false, bh, (short)0, c, false, false);
                c = __builtin_amdgcn_wmma_f32_16x16x32_bf16(
                        false, ah, false, bl, (short)0, c, false, false);
                c = __builtin_amdgcn_wmma_f32_16x16x32_bf16(
                        false, al, false, bh, (short)0, c, false, false);
            }
        }

        const int vcol = v0 + wave * 16 + ln;
        const bool valid = vcol < NUM_ITEMS;
#pragma unroll
        for (int r = 0; r < 8; ++r) {
            float x = valid ? c[r] : NEG_INF;
            float m = x;
#pragma unroll
            for (int msk = 1; msk < 16; msk <<= 1)
                m = fmaxf(m, __shfl_xor(m, msk, 32));
            if (m > NEG_INF) {
                float e = __expf(x - m);
#pragma unroll
                for (int msk = 1; msk < 16; msk <<= 1)
                    e += __shfl_xor(e, msk, 32);
                float Mo = Mrun[r];
                float Mn = fmaxf(Mo, m);
                Srun[r] = Srun[r] * __expf(Mo - Mn) + e * __expf(m - Mn);
                Mrun[r] = Mn;
            }
        }
    }

    if (ln == 0) {
#pragma unroll
        for (int r = 0; r < 8; ++r) {
            float2 p; p.x = Mrun[r]; p.y = Srun[r];
            comb[wave * 16 + (r + 8 * half)] = p;
        }
    }
    __syncthreads();
    if (t < 16) {
        float M = NEG_INF, S = 0.0f;
#pragma unroll
        for (int w = 0; w < 8; ++w) {
            float2 p = comb[w * 16 + t];
            if (p.x > NEG_INF) {
                float Mn = fmaxf(M, p.x);
                S = S * __expf(M - Mn) + p.y * __expf(p.x - Mn);
                M = Mn;
            }
        }
        float2 o; o.x = M; o.y = S;
        ws[(size_t)(rowBase + t) * vblocks + blockIdx.y] = o;
    }
}

// ---------------------------------------------------------------------------
// Kernel 2: per-row label logit in exact fp32: dot(hidden[row,:], item_emb[lab,:])
// ---------------------------------------------------------------------------
__global__ __launch_bounds__(256)
void label_logit_kernel(const float* __restrict__ hidden,
                        const float* __restrict__ item_emb,
                        const int* __restrict__ labels,
                        float* __restrict__ out)
{
    __shared__ float red[256];
    const int row = blockIdx.x;          // 0..1023
    const int b = row >> 7, l = row & 127;
    if (l == 127) { if (threadIdx.x == 0) out[row] = 0.0f; return; }

    int lab = labels[b * L_MAIN + l + 1] - LABEL_OFF;
    lab = lab < 0 ? 0 : (lab > NUM_ITEMS - 1 ? NUM_ITEMS - 1 : lab);

    const float* h = hidden + (size_t)row * D_DIM;
    const float* e = item_emb + (size_t)lab * D_DIM;
    float acc = 0.0f;
#pragma unroll
    for (int j = 0; j < 3; ++j) {
        int d = threadIdx.x + j * 256;
        acc += h[d] * e[d];
    }
    red[threadIdx.x] = acc;
    __syncthreads();
    for (int s = 128; s > 0; s >>= 1) {
        if (threadIdx.x < s) red[threadIdx.x] += red[threadIdx.x + s];
        __syncthreads();
    }
    if (threadIdx.x == 0) out[row] = red[0];
}

// ---------------------------------------------------------------------------
// Kernel 3: merge slice partials, apply mask, produce scalar loss
// ---------------------------------------------------------------------------
__global__ __launch_bounds__(256)
void finalize_kernel(const float2* __restrict__ ws,
                     const float* __restrict__ lab_logit,
                     const int* __restrict__ amask,
                     const int* __restrict__ prompt_len_p,
                     float* __restrict__ out, int vblocks)
{
    __shared__ float rn[256], rw[256];
    const int pl = *prompt_len_p;
    float sum_nll = 0.0f, sum_w = 0.0f;

    for (int pos = threadIdx.x; pos < B_SZ * (L_MAIN - 1); pos += 256) {
        const int b = pos / (L_MAIN - 1);
        const int l = pos % (L_MAIN - 1);
        if (amask[b * L_TOT + pl + 1 + l] == 1) {
            const int row = b * L_MAIN + l;
            float M = NEG_INF, S = 0.0f;
            for (int cidx = 0; cidx < vblocks; ++cidx) {
                float2 p = ws[(size_t)row * vblocks + cidx];
                if (p.x > NEG_INF && p.y > 0.0f) {
                    float Mn = fmaxf(M, p.x);
                    S = S * __expf(M - Mn) + p.y * __expf(p.x - Mn);
                    M = Mn;
                }
            }
            sum_nll += M + __logf(S) - lab_logit[row];
            sum_w += 1.0f;
        }
    }
    rn[threadIdx.x] = sum_nll;
    rw[threadIdx.x] = sum_w;
    __syncthreads();
    for (int s = 128; s > 0; s >>= 1) {
        if (threadIdx.x < s) {
            rn[threadIdx.x] += rn[threadIdx.x + s];
            rw[threadIdx.x] += rw[threadIdx.x + s];
        }
        __syncthreads();
    }
    if (threadIdx.x == 0) out[0] = rn[0] / rw[0];
}

// ---------------------------------------------------------------------------
static inline size_t align256(size_t x) { return (x + 255) & ~(size_t)255; }

extern "C" void kernel_launch(void* const* d_in, const int* in_sizes, int n_in,
                              void* d_out, int out_size, void* d_ws, size_t ws_size,
                              hipStream_t stream)
{
    const float* hidden   = (const float*)d_in[0];   // (8,128,768) f32
    const float* item_emb = (const float*)d_in[1];   // (100000,768) f32
    const int*   labels   = (const int*)d_in[2];     // (8,128)
    const int*   amask    = (const int*)d_in[3];     // (8,160)
    const int*   plen     = (const int*)d_in[4];     // scalar on device
    float*       out      = (float*)d_out;

    // ---- workspace layout ----
    int VB = 96;
    const size_t per_slice = (size_t)NROWS * sizeof(float2);  // 8 KB / slice
    if (ws_size < (size_t)VB * per_slice + NROWS * sizeof(float)) {
        size_t fit = (ws_size > NROWS * sizeof(float))
                   ? (ws_size - NROWS * sizeof(float)) / per_slice : 1;
        VB = fit < 1 ? 1 : (int)fit;
    }
    const size_t hid_plane  = (size_t)NROWS * (D_DIM / 2) * sizeof(unsigned);       // 1.5 MB
    const size_t item_plane = (size_t)NPAD_ITEMS * (D_DIM / 2) * sizeof(unsigned);  // 153.7 MB

    size_t off_part = 0;
    size_t off_lab  = align256(off_part + (size_t)VB * per_slice);
    size_t off_hHi  = align256(off_lab + NROWS * sizeof(float));
    size_t off_hLo  = off_hHi + hid_plane;
    size_t off_iHi  = off_hLo + hid_plane;
    size_t off_iLo  = off_iHi + item_plane;
    size_t total    = off_iLo + item_plane;

    float2*   ws_ms  = (float2*)d_ws;
    float*    ws_lab = (float*)((char*)d_ws + off_lab);
    unsigned* hidHi  = (unsigned*)((char*)d_ws + off_hHi);
    unsigned* hidLo  = (unsigned*)((char*)d_ws + off_hLo);
    unsigned* itemHi = (unsigned*)((char*)d_ws + off_iHi);
    unsigned* itemLo = (unsigned*)((char*)d_ws + off_iLo);

    const bool presplit = ws_size >= total;

    dim3 g1(NROWS / 16, VB);
    if (presplit) {
        split_planes_kernel<<<512, 256, 0, stream>>>(
            hidden, hidHi, hidLo,
            (long)NROWS * (D_DIM / 2), (long)NROWS * (D_DIM / 2));
        split_planes_kernel<<<8192, 256, 0, stream>>>(
            item_emb, itemHi, itemLo,
            (long)NUM_ITEMS * (D_DIM / 2), (long)NPAD_ITEMS * (D_DIM / 2));
        ce_gemm_softmax_presplit<<<g1, 256, SMEM_BYTES, stream>>>(
            hidHi, hidLo, itemHi, itemLo, ws_ms, VB);
    } else {
        ce_gemm_softmax_fallback<<<g1, 256, 0, stream>>>(
            hidden, item_emb, ws_ms, VB);
    }
    label_logit_kernel<<<NROWS, 256, 0, stream>>>(hidden, item_emb, labels, ws_lab);
    finalize_kernel<<<1, 256, 0, stream>>>(ws_ms, ws_lab, amask, plen, out, VB);
}
